// HybridQuanvolutionNet_65481071400496
// MI455X (gfx1250) — matrix operand order repaired and checked by reading, not compiled, not run on hardware
//
#include <hip/hip_runtime.h>
#include <hip/hip_bf16.h>
#include <math.h>

// ---------------------------------------------------------------------------
// HybridQuanvolutionNet fused pipeline for gfx1250 (MI455X)
//   stage 1: quanv collapses to per-pixel cos(sigmoid(s*x)+p) products (f16 out)
//   stage 2: 784->128->64->32 MLP via v_wmma_f32_16x16x32_f16 (f32 accum),
//            weights cooperatively staged in LDS and shared by all 8 waves/WG
//   stage 3: fidelity chain scan (single block), atomic segment sums, classify
// ---------------------------------------------------------------------------

#define Bsz   8192
#define GRID  14           // 28/2 patches per side
#define K1P   800          // 784 padded to multiple of 32
#define LD1   136          // 128 + 8 halves pad (16B multiple, avoids bank conflicts)
#define LD2   72           // 64 + 8 halves pad
#define WLD1  96           // LDS stride for staged W1 chunk (3 kt * 32 halves)

typedef __attribute__((ext_vector_type(16))) _Float16 v16h;
typedef __attribute__((ext_vector_type(8)))  _Float16 v8h;
typedef __attribute__((ext_vector_type(8)))  float    v8f;

static __device__ inline v16h mk16(v8h lo, v8h hi) {
    return __builtin_shufflevector(lo, hi, 0,1,2,3,4,5,6,7,8,9,10,11,12,13,14,15);
}

// ---------------- weight conversion: f32 -> f16, W1 K-padded to 800 --------
__global__ void prep_weights_kernel(const float* __restrict__ W1,
                                    const float* __restrict__ W2,
                                    const float* __restrict__ W3,
                                    _Float16* __restrict__ W1h,
                                    _Float16* __restrict__ W2h,
                                    _Float16* __restrict__ W3h) {
    int t = blockIdx.x * 256 + threadIdx.x;
    const int n1 = 128 * K1P, n2 = 64 * 128, n3 = 32 * 64;
    if (t < n1) {
        int r = t / K1P, k = t % K1P;
        W1h[t] = (_Float16)((k < 784) ? W1[r * 784 + k] : 0.0f);
    } else if (t < n1 + n2) {
        int u = t - n1;
        W2h[u] = (_Float16)W2[u];
    } else if (t < n1 + n2 + n3) {
        int u = t - n1 - n2;
        W3h[u] = (_Float16)W3[u];
    }
}

// ---------------- quanv: closed-form circuit, writes q[B][800] f16 ---------
__global__ void quanv_kernel(const float* __restrict__ x,
                             const float* __restrict__ conv_w,
                             const float* __restrict__ qp,
                             _Float16* __restrict__ qh) {
    int tid = blockIdx.x * 256 + threadIdx.x;
    int n = tid / 200;              // image
    int c = tid % 200;              // 196 patches + 4 zero-pad groups
    if (n >= Bsz) return;
    _Float16* row = qh + (size_t)n * K1P;
    if (c >= 196) {
        int f = c * 4;              // features 784..799 -> zero pad
        row[f + 0] = (_Float16)0.0f; row[f + 1] = (_Float16)0.0f;
        row[f + 2] = (_Float16)0.0f; row[f + 3] = (_Float16)0.0f;
        return;
    }
    int i = c / GRID, j = c % GRID;
    const float* xb = x + (size_t)n * 784 + (2 * i) * 28 + 2 * j;
    float s = conv_w[0];
    float px0 = xb[0], px1 = xb[1], px2 = xb[28], px3 = xb[29];
    float m0 = cosf(1.0f / (1.0f + expf(-s * px0)) + qp[0]);
    float m1 = cosf(1.0f / (1.0f + expf(-s * px1)) + qp[1]);
    float m2 = cosf(1.0f / (1.0f + expf(-s * px2)) + qp[2]);
    float m3 = cosf(1.0f / (1.0f + expf(-s * px3)) + qp[3]);
    // CNOT ring => XOR parities => products of single-wire expectations
    float o1 = m0 * m1;
    float o2 = o1 * m2;
    float o3 = o2 * m3;
    float o0 = m1 * m2 * m3;
    int f = c * 4;
    row[f + 0] = (_Float16)o0; row[f + 1] = (_Float16)o1;
    row[f + 2] = (_Float16)o2; row[f + 3] = (_Float16)o3;
}

// ---------------- fused MLP: q->h1(relu)->h2(relu)->lat, all WMMA ----------
// 8 waves/block, 16 rows/wave; B-operands staged in LDS once per block.
__global__ __launch_bounds__(256) void mlp_kernel(
        const _Float16* __restrict__ qh,
        const _Float16* __restrict__ W1h,
        const _Float16* __restrict__ W2h,
        const _Float16* __restrict__ W3h,
        const float* __restrict__ b1,
        const float* __restrict__ b2,
        const float* __restrict__ b3,
        float* __restrict__ lat) {
    // wstage: 128 rows x 96 halves = 24576B (W1 K-chunks; later W2/W3 padded)
    __shared__ __align__(16) _Float16 wstage[128 * WLD1];
    // per-wave activation tiles; h2 reuses h1 region (16*LD2 <= 16*LD1)
    __shared__ __align__(16) _Float16 h1s[8][16 * LD1];

    const int tid = threadIdx.x;
    const int wave = tid >> 5;
    const int lane = tid & 31;
    const int laneHi = lane >> 4;        // K half-select
    const int nlo = lane & 15;           // N column / A row within tile
    const int row0 = (blockIdx.x * 8 + wave) * 16;

    // ---- GEMM1: [16,800] x [800,128] -> h1, K in chunks of 3*32 ----
    v8f acc[8] = {};
    const _Float16* aBase = qh + (size_t)(row0 + nlo) * K1P + laneHi * 8;
    for (int kt0 = 0; kt0 < 25; kt0 += 3) {
        const int kc = (25 - kt0 < 3) ? (25 - kt0) : 3;
        // cooperative stage of W1h[:, kt0*32 .. kt0*32+kc*32) into LDS
        if (kc == 3) {
            for (int u = tid; u < 128 * 12; u += 256) {
                int row = u / 12, cu = u - row * 12;
                *(v8h*)(wstage + row * WLD1 + cu * 8) =
                    *(const v8h*)(W1h + (size_t)row * K1P + kt0 * 32 + cu * 8);
            }
        } else {
            for (int u = tid; u < 128 * 4; u += 256) {
                int row = u >> 2, cu = u & 3;
                *(v8h*)(wstage + row * WLD1 + cu * 8) =
                    *(const v8h*)(W1h + (size_t)row * K1P + kt0 * 32 + cu * 8);
            }
        }
        // prefetch next A chunk while staging
        if (kt0 + 3 < 25) __builtin_prefetch(aBase + (kt0 + 3) * 32, 0, 0);
        __syncthreads();
        for (int kt = 0; kt < kc; ++kt) {
            const int k0 = (kt0 + kt) * 32;
            v16h a = mk16(*(const v8h*)(aBase + k0), *(const v8h*)(aBase + k0 + 16));
            #pragma unroll
            for (int nt = 0; nt < 8; ++nt) {
                const _Float16* bp = wstage + (nt * 16 + nlo) * WLD1 + kt * 32 + laneHi * 8;
                v16h b = mk16(*(const v8h*)bp, *(const v8h*)(bp + 16));
                acc[nt] = __builtin_amdgcn_wmma_f32_16x16x32_f16(
                    false, a, false, b, (short)0, acc[nt], false, false);
            }
        }
        __syncthreads();
    }
    // bias + relu -> h1 tile in LDS
    _Float16* h1 = h1s[wave];
    #pragma unroll
    for (int nt = 0; nt < 8; ++nt) {
        float bias = b1[nt * 16 + nlo];
        #pragma unroll
        for (int r = 0; r < 8; ++r) {
            int m = r + 8 * laneHi;
            float v = acc[nt][r] + bias;
            h1[m * LD1 + nt * 16 + nlo] = (_Float16)(v > 0.0f ? v : 0.0f);
        }
    }
    // stage W2 (stride 136) and W3 (stride 72) into wstage, bank-conflict padded
    _Float16* w2s = wstage;               // 64 rows x LD1 stride
    _Float16* w3s = wstage + 64 * LD1;    // 32 rows x LD2 stride
    for (int u = tid; u < 64 * 16; u += 256) {
        int row = u >> 4, cu = u & 15;
        *(v8h*)(w2s + row * LD1 + cu * 8) = *(const v8h*)(W2h + row * 128 + cu * 8);
    }
    for (int u = tid; u < 32 * 8; u += 256) {
        int row = u >> 3, cu = u & 7;
        *(v8h*)(w3s + row * LD2 + cu * 8) = *(const v8h*)(W3h + row * 64 + cu * 8);
    }
    __syncthreads();

    // ---- GEMM2: [16,128] x [128,64] -> h2 ----
    v8f acc2[4] = {};
    const _Float16* a2 = h1 + nlo * LD1 + laneHi * 8;
    #pragma unroll
    for (int kt = 0; kt < 4; ++kt) {
        const int k0 = kt * 32;
        v16h a = mk16(*(const v8h*)(a2 + k0), *(const v8h*)(a2 + k0 + 16));
        #pragma unroll
        for (int nt = 0; nt < 4; ++nt) {
            const _Float16* bp = w2s + (nt * 16 + nlo) * LD1 + k0 + laneHi * 8;
            v16h b = mk16(*(const v8h*)bp, *(const v8h*)(bp + 16));
            acc2[nt] = __builtin_amdgcn_wmma_f32_16x16x32_f16(
                false, a, false, b, (short)0, acc2[nt], false, false);
        }
    }
    __syncthreads();   // everyone done reading h1 before overwriting region with h2
    _Float16* h2 = h1s[wave];             // reuse h1 region, stride LD2
    #pragma unroll
    for (int nt = 0; nt < 4; ++nt) {
        float bias = b2[nt * 16 + nlo];
        #pragma unroll
        for (int r = 0; r < 8; ++r) {
            int m = r + 8 * laneHi;
            float v = acc2[nt][r] + bias;
            h2[m * LD2 + nt * 16 + nlo] = (_Float16)(v > 0.0f ? v : 0.0f);
        }
    }
    __syncthreads();

    // ---- GEMM3: [16,64] x [64,32] -> lat (f32 global) ----
    v8f acc3[2] = {};
    const _Float16* a3 = h2 + nlo * LD2 + laneHi * 8;
    #pragma unroll
    for (int kt = 0; kt < 2; ++kt) {
        const int k0 = kt * 32;
        v16h a = mk16(*(const v8h*)(a3 + k0), *(const v8h*)(a3 + k0 + 16));
        #pragma unroll
        for (int nt = 0; nt < 2; ++nt) {
            const _Float16* bp = w3s + (nt * 16 + nlo) * LD2 + k0 + laneHi * 8;
            v16h b = mk16(*(const v8h*)bp, *(const v8h*)(bp + 16));
            acc3[nt] = __builtin_amdgcn_wmma_f32_16x16x32_f16(
                false, a, false, b, (short)0, acc3[nt], false, false);
        }
    }
    #pragma unroll
    for (int nt = 0; nt < 2; ++nt) {
        float bias = b3[nt * 16 + nlo];
        #pragma unroll
        for (int r = 0; r < 8; ++r) {
            int m = r + 8 * laneHi;
            lat[(size_t)(row0 + m) * 32 + nt * 16 + nlo] = acc3[nt][r] + bias;
        }
    }
}

// ---------------- fidelity chain: norms, breaks, cumsum -> seg -------------
__global__ __launch_bounds__(1024) void seg_scan_kernel(
        const float* __restrict__ lat,
        float* __restrict__ norms,
        int* __restrict__ brk,
        int* __restrict__ seg) {
    const int tid = threadIdx.x;
    for (int i = tid; i < Bsz; i += 1024) {
        float s = 0.0f;
        #pragma unroll
        for (int k = 0; k < 32; ++k) { float v = lat[i * 32 + k]; s += v * v; }
        norms[i] = sqrtf(s);
    }
    __syncthreads();
    for (int i = tid; i < Bsz - 1; i += 1024) {
        float d = 0.0f;
        #pragma unroll
        for (int k = 0; k < 32; ++k) d += lat[i * 32 + k] * lat[(i + 1) * 32 + k];
        float c = d / ((norms[i] + 1e-12f) * (norms[i + 1] + 1e-12f));
        brk[i] = (c * c < 0.5f) ? 1 : 0;
    }
    __syncthreads();
    // block-wide inclusive scan in chunks of 1024
    __shared__ int sh[1024];
    __shared__ int carry;
    if (tid == 0) { carry = 0; seg[0] = 0; }
    __syncthreads();
    for (int base = 0; base < Bsz - 1; base += 1024) {
        int i = base + tid;
        sh[tid] = (i < Bsz - 1) ? brk[i] : 0;
        __syncthreads();
        for (int off = 1; off < 1024; off <<= 1) {
            int t = (tid >= off) ? sh[tid - off] : 0;
            __syncthreads();
            sh[tid] += t;
            __syncthreads();
        }
        if (i < Bsz - 1) seg[i + 1] = carry + sh[tid];
        __syncthreads();
        if (tid == 1023) carry += sh[1023];
        __syncthreads();
    }
}

// ---------------- zero segment accumulators --------------------------------
__global__ void zero_kernel(float* __restrict__ sums, float* __restrict__ cnt) {
    int t = blockIdx.x * 256 + threadIdx.x;
    if (t < Bsz * 32) sums[t] = 0.0f;
    if (t < Bsz) cnt[t] = 0.0f;
}

// ---------------- segment sums via global atomics --------------------------
__global__ void segsum_kernel(const float* __restrict__ lat,
                              const int* __restrict__ seg,
                              float* __restrict__ sums,
                              float* __restrict__ cnt) {
    int i = blockIdx.x * 256 + threadIdx.x;
    if (i >= Bsz) return;
    int s = seg[i];
    #pragma unroll
    for (int k = 0; k < 32; ++k)
        atomicAdd(&sums[s * 32 + k], lat[i * 32 + k]);
    atomicAdd(&cnt[s], 1.0f);
}

// ---------------- centroid classifier + masking ----------------------------
__global__ void cls_kernel(const float* __restrict__ sums,
                           const float* __restrict__ cnt,
                           const float* __restrict__ cls_w,
                           const float* __restrict__ cls_b,
                           float* __restrict__ out) {
    int r = blockIdx.x * 256 + threadIdx.x;
    if (r >= Bsz) return;
    float c = cnt[r];
    if (c > 0.0f) {
        float inv = 1.0f / fmaxf(c, 1.0f);
        float cent[32];
        #pragma unroll
        for (int k = 0; k < 32; ++k) cent[k] = sums[r * 32 + k] * inv;
        #pragma unroll
        for (int j = 0; j < 10; ++j) {
            float acc = cls_b[j];
            #pragma unroll
            for (int k = 0; k < 32; ++k) acc += cent[k] * cls_w[j * 32 + k];
            out[r * 10 + j] = acc;
        }
    } else {
        #pragma unroll
        for (int j = 0; j < 10; ++j) out[r * 10 + j] = 0.0f;
    }
}

// ---------------- workspace layout (all 256B aligned) ----------------------
static constexpr size_t OFF_W1H   = 0;                         // 128*800*2 = 204800
static constexpr size_t OFF_W2H   = 204800;                    // 64*128*2  = 16384
static constexpr size_t OFF_W3H   = 221184;                    // 32*64*2   = 4096
static constexpr size_t OFF_QH    = 225280;                    // 8192*800*2 = 13107200
static constexpr size_t OFF_LAT   = 13332480;                  // 8192*32*4  = 1048576
static constexpr size_t OFF_NORMS = 14381056;                  // 8192*4
static constexpr size_t OFF_BRK   = 14413824;                  // 8192*4
static constexpr size_t OFF_SEG   = 14446592;                  // 8192*4
static constexpr size_t OFF_SUMS  = 14479360;                  // 8192*32*4
static constexpr size_t OFF_CNT   = 15527936;                  // 8192*4

extern "C" void kernel_launch(void* const* d_in, const int* in_sizes, int n_in,
                              void* d_out, int out_size, void* d_ws, size_t ws_size,
                              hipStream_t stream) {
    const float* x      = (const float*)d_in[0];
    const float* conv_w = (const float*)d_in[1];
    const float* qp     = (const float*)d_in[2];
    const float* W1     = (const float*)d_in[3];
    const float* b1     = (const float*)d_in[4];
    const float* W2     = (const float*)d_in[5];
    const float* b2     = (const float*)d_in[6];
    const float* W3     = (const float*)d_in[7];
    const float* b3     = (const float*)d_in[8];
    const float* clsw   = (const float*)d_in[9];
    const float* clsb   = (const float*)d_in[10];
    float* out = (float*)d_out;

    char* ws = (char*)d_ws;
    _Float16* W1h  = (_Float16*)(ws + OFF_W1H);
    _Float16* W2h  = (_Float16*)(ws + OFF_W2H);
    _Float16* W3h  = (_Float16*)(ws + OFF_W3H);
    _Float16* qh   = (_Float16*)(ws + OFF_QH);
    float*    lat  = (float*)(ws + OFF_LAT);
    float*    nrm  = (float*)(ws + OFF_NORMS);
    int*      brk  = (int*)(ws + OFF_BRK);
    int*      seg  = (int*)(ws + OFF_SEG);
    float*    sums = (float*)(ws + OFF_SUMS);
    float*    cnt  = (float*)(ws + OFF_CNT);

    // 1) weights -> f16 (W1 K-padded)
    {
        int total = 128 * K1P + 64 * 128 + 32 * 64;
        prep_weights_kernel<<<(total + 255) / 256, 256, 0, stream>>>(
            W1, W2, W3, W1h, W2h, W3h);
    }
    // 2) quanv (closed form) -> q f16 [8192, 800]
    quanv_kernel<<<(Bsz * 200) / 256, 256, 0, stream>>>(x, conv_w, qp, qh);
    // 3) fused WMMA MLP (LDS-staged weights) -> lat [8192, 32]
    mlp_kernel<<<Bsz / 128, 256, 0, stream>>>(qh, W1h, W2h, W3h, b1, b2, b3, lat);
    // 4) fidelity chain scan -> seg labels
    seg_scan_kernel<<<1, 1024, 0, stream>>>(lat, nrm, brk, seg);
    // 5) zero accumulators, 6) atomic segment sums
    zero_kernel<<<(Bsz * 32 + 255) / 256, 256, 0, stream>>>(sums, cnt);
    segsum_kernel<<<(Bsz + 255) / 256, 256, 0, stream>>>(lat, seg, sums, cnt);
    // 7) centroid classifier with empty-row masking
    cls_kernel<<<(Bsz + 255) / 256, 256, 0, stream>>>(sums, cnt, clsw, clsb, out);
}
